// VectorQuantizer_1005022347700
// MI455X (gfx1250) — matrix operand order repaired and checked by reading, not compiled, not run on hardware
//
#include <hip/hip_runtime.h>

// ---------------------------------------------------------------------------
// VQ-VAE vector quantizer for MI455X (gfx1250, wave32, WMMA).
//
// x:          [16, 256, 64, 64] f32  (N = 16*64*64 = 65536 rows, C = 256)
// embeddings: [1024, 256] f32        (K = 1024 codes)
//
// d_out (floats, concatenated in reference return order):
//   [0, 16777216)          quantized_out [B,C,H,W] f32
//   [16777216]             total_loss (scalar f32)
//   [16777217, +65536)     idx [B, H*W] as int32 bit patterns
//   [16842753, +262144)    embeddings pass-through f32
//
// Design: bf16 WMMA computes x.e (f32 accumulate); argmin discriminant is
// ||e||^2 - 2 x.e (the ||x||^2 row constant is irrelevant for argmin).
// Each wave keeps TWO 16-row bf16 A sets resident (32 rows/wave) and issues
// two v_wmma_f32_16x16x32_bf16 per B fragment; B fragments are double-
// buffered in registers so the next 32B load overlaps the current WMMA pair.
// ---------------------------------------------------------------------------

typedef __attribute__((ext_vector_type(16))) __bf16 v16bf;
typedef __attribute__((ext_vector_type(8)))  float  v8f;

#define N_ROWS   65536
#define K_CODES  1024
#define C_DIM    256
#define HWSZ     4096        // H*W, stride (in floats) between c planes
#define PLANE    1048576     // 256*4096 floats per batch image

__device__ __forceinline__ unsigned short f2bf(float f) {
    // round-to-nearest-even float -> bf16 (inputs are finite normals here)
    unsigned u = __float_as_uint(f);
    unsigned r = u + 0x7FFFu + ((u >> 16) & 1u);
    return (unsigned short)(r >> 16);
}

union AFrag { v16bf v; unsigned short u[16]; };
union BFrag { v16bf v; uint4 q[2]; };

// ---------------------------------------------------------------------------
// Prep: embeddings f32 -> bf16 table (ws), per-code squared norms (ws),
//       pass-through copy to d_out, zero the loss slot.
// grid = 1024 blocks (one code per block), block = 256 threads.
// ---------------------------------------------------------------------------
__global__ __launch_bounds__(256) void vq_prep(const float* __restrict__ E,
                                               unsigned short* __restrict__ ebf,
                                               float* __restrict__ enorm,
                                               float* __restrict__ outE,
                                               float* __restrict__ lossSlot) {
    const int k = blockIdx.x;
    const int t = threadIdx.x;
    const float v = E[k * C_DIM + t];
    ebf [k * C_DIM + t] = f2bf(v);
    outE[k * C_DIM + t] = v;

    float s = v * v;
    #pragma unroll
    for (int m = 16; m >= 1; m >>= 1) s += __shfl_xor(s, m, 32);

    __shared__ float red[8];
    if ((t & 31) == 0) red[t >> 5] = s;
    __syncthreads();
    if (t == 0) {
        float tot = 0.f;
        #pragma unroll
        for (int i = 0; i < 8; ++i) tot += red[i];
        enorm[k] = tot;
        if (k == 0) *lossSlot = 0.f;
    }
}

// ---------------------------------------------------------------------------
// Main: 32 rows per wave (two A sets), 256 rows per block, WMMA sweep over
// all 1024 codes with double-buffered B fragments, fused argmin, then
// per-thread quantized gather/scatter with coalesced stores + loss.
// grid = 256 blocks, block = 256 threads (8 waves).
// ---------------------------------------------------------------------------
__global__ __launch_bounds__(256, 1) void vq_main(const float* __restrict__ X,
                                                  const float* __restrict__ E,
                                                  const unsigned short* __restrict__ ebf,
                                                  const float* __restrict__ enorm,
                                                  float* __restrict__ outQ,
                                                  float* __restrict__ lossSlot,
                                                  int* __restrict__ idxOut) {
    __shared__ int   sIdx[256];   // winning code per block-local row
    __shared__ float sRed[8];

    const int tid  = threadIdx.x;
    const int wave = tid >> 5;
    const int lane = tid & 31;
    const int lo   = lane & 15;        // row (A/C) or column (B) selector
    const int hi   = lane >> 4;        // which 16-lane half
    const int hi8  = hi * 8;

    const int    rowbase = (blockIdx.x * 8 + wave) * 32;   // 32 consecutive n
    const size_t b       = (size_t)(rowbase >> 12);        // batch image
    const int    inb     = (rowbase & (HWSZ - 1)) + lo;    // h*64 + w

    // ---- Load & convert this wave's 32 rows of x into two A-fragment sets.
    // A k order used here (contiguous, same as B): lane m (0-15): K {c0..c0+15};
    // lane m+16: K {c0+16..c0+31}.  x[m,c] lives at stride 4096.
    const float* xb0 = X + b * PLANE + inb;        // rows rowbase..+15
    const float* xb1 = xb0 + 16;                   // rows rowbase+16..+31
    AFrag A0[8], A1[8];
    #pragma unroll
    for (int j = 0; j < 8; ++j) {
        const int c0 = 32 * j;
        #pragma unroll
        for (int t = 0; t < 8; ++t) {
            const size_t oL = (size_t)(c0 + hi * 16 + t)     * HWSZ;
            const size_t oH = (size_t)(c0 + hi * 16 + 8 + t) * HWSZ;
            A0[j].u[t]     = f2bf(xb0[oL]);
            A0[j].u[8 + t] = f2bf(xb0[oH]);
            A1[j].u[t]     = f2bf(xb1[oL]);
            A1[j].u[8 + t] = f2bf(xb1[oH]);
        }
    }

    // ---- Sweep all 1024 codes in 16-column tiles -------------------------
    float minv0[8], minv1[8];
    int   mini0[8], mini1[8];
    #pragma unroll
    for (int r = 0; r < 8; ++r) {
        minv0[r] = 3.4e38f; mini0[r] = 0;
        minv1[r] = 3.4e38f; mini1[r] = 0;
    }

    for (int colbase = 0; colbase < K_CODES; colbase += 16) {
        // B k order used here (contiguous, same as A): lane n (0-15) holds e[colbase+n, c0..c0+15],
        // lane n+16 holds e[colbase+n, c0+16..c0+31] -- contiguous 32B loads.
        const unsigned short* bp = ebf + (size_t)(colbase + lo) * C_DIM + hi * 16;
        if (colbase + 16 < K_CODES)
            __builtin_prefetch(bp + 16 * C_DIM, 0, 0);   // next tile's B rows

        // Double-buffered B fragments: load j+1 while WMMAs consume j.
        BFrag Bb[2];
        Bb[0].q[0] = *(const uint4*)(bp);
        Bb[0].q[1] = *(const uint4*)(bp + 8);

        v8f acc0 = {0.f, 0.f, 0.f, 0.f, 0.f, 0.f, 0.f, 0.f};
        v8f acc1 = {0.f, 0.f, 0.f, 0.f, 0.f, 0.f, 0.f, 0.f};
        #pragma unroll
        for (int j = 0; j < 8; ++j) {
            if (j < 7) {
                Bb[(j + 1) & 1].q[0] = *(const uint4*)(bp + 32 * (j + 1));
                Bb[(j + 1) & 1].q[1] = *(const uint4*)(bp + 32 * (j + 1) + 8);
            }
            acc0 = __builtin_amdgcn_wmma_f32_16x16x32_bf16(
                false, A0[j].v, false, Bb[j & 1].v, (short)0, acc0, false, false);
            acc1 = __builtin_amdgcn_wmma_f32_16x16x32_bf16(
                false, A1[j].v, false, Bb[j & 1].v, (short)0, acc1, false, false);
        }

        const float en  = enorm[colbase + lo];
        const int   col = colbase + lo;
        #pragma unroll
        for (int r = 0; r < 8; ++r) {
            const float s0 = fmaf(-2.0f, acc0[r], en);   // ||e||^2 - 2 x.e
            const float s1 = fmaf(-2.0f, acc1[r], en);
            if (s0 < minv0[r]) { minv0[r] = s0; mini0[r] = col; }
            if (s1 < minv1[r]) { minv1[r] = s1; mini1[r] = col; }
        }
    }

    // ---- Argmin across the 16 lanes sharing each row (xor <= 8 stays in half)
    #pragma unroll
    for (int r = 0; r < 8; ++r) {
        #pragma unroll
        for (int s = 8; s >= 1; s >>= 1) {
            float ov0 = __shfl_xor(minv0[r], s, 32);
            int   oi0 = __shfl_xor(mini0[r], s, 32);
            if (ov0 < minv0[r] || (ov0 == minv0[r] && oi0 < mini0[r])) {
                minv0[r] = ov0; mini0[r] = oi0;
            }
            float ov1 = __shfl_xor(minv1[r], s, 32);
            int   oi1 = __shfl_xor(mini1[r], s, 32);
            if (ov1 < minv1[r] || (ov1 == minv1[r] && oi1 < mini1[r])) {
                minv1[r] = ov1; mini1[r] = oi1;
            }
        }
    }

    // C/D layout: VGPR r holds row r (lanes 0-15) / row 8+r (lanes 16-31).
    if (lo == 0) {
        #pragma unroll
        for (int r = 0; r < 8; ++r) {
            const int row0 = r + hi8;          // set 0: rows rowbase..+15
            const int row1 = 16 + r + hi8;     // set 1: rows rowbase+16..+31
            sIdx[wave * 32 + row0] = mini0[r];
            sIdx[wave * 32 + row1] = mini1[r];
            idxOut[rowbase + row0] = mini0[r];
            idxOut[rowbase + row1] = mini1[r];
        }
    }
    __syncthreads();

    // ---- Per-thread quantized output + loss ------------------------------
    // 256 consecutive rows per block -> 1KB contiguous stores per c plane.
    const int    nrow = blockIdx.x * 256 + tid;
    const size_t bb   = (size_t)(nrow >> 12);
    const int    inb2 = nrow & (HWSZ - 1);
    const int    code = sIdx[tid];
    const float* er   = E + (size_t)code * C_DIM;
    const float* xp   = X    + bb * PLANE + inb2;
    float*       op   = outQ + bb * PLANE + inb2;

    float lsum = 0.f;
    #pragma unroll 4
    for (int c = 0; c < C_DIM; ++c) {
        const float q  = er[c];
        const float xv = xp[(size_t)c * HWSZ];
        op[(size_t)c * HWSZ] = q;
        const float d = q - xv;
        lsum = fmaf(d, d, lsum);
    }

    #pragma unroll
    for (int s = 16; s >= 1; s >>= 1) lsum += __shfl_xor(lsum, s, 32);
    if ((tid & 31) == 0) sRed[tid >> 5] = lsum;
    __syncthreads();
    if (tid == 0) {
        float tot = 0.f;
        #pragma unroll
        for (int i = 0; i < 8; ++i) tot += sRed[i];
        // total = (1 + 0.25) * mean((q - x)^2) over N*C = 16777216 elements
        atomicAdd(lossSlot, tot * (1.25f / 16777216.f));
    }
}

// ---------------------------------------------------------------------------
extern "C" void kernel_launch(void* const* d_in, const int* in_sizes, int n_in,
                              void* d_out, int out_size, void* d_ws, size_t ws_size,
                              hipStream_t stream) {
    const float* X = (const float*)d_in[0];   // [16,256,64,64] f32
    const float* E = (const float*)d_in[1];   // [1024,256] f32

    float* out      = (float*)d_out;
    float* outQ     = out;                          // 16777216 floats
    float* lossSlot = out + 16777216;               // 1 float
    int*   idxOut   = (int*)(out + 16777217);       // 65536 int32
    float* outE     = out + 16777217 + 65536;       // 262144 floats

    unsigned short* ebf   = (unsigned short*)d_ws;                       // 512 KB
    float*          enorm = (float*)((char*)d_ws + K_CODES * C_DIM * 2); // 4 KB

    vq_prep<<<K_CODES, 256, 0, stream>>>(E, ebf, enorm, outE, lossSlot);
    vq_main<<<N_ROWS / 256, 256, 0, stream>>>(X, E, ebf, enorm, outQ, lossSlot, idxOut);
}
